// MultiHeadAttentionQuantum_65481071402175
// MI455X (gfx1250) — matrix operand order repaired and checked by reading, not compiled, not run on hardware
//
#include <hip/hip_runtime.h>
#include <hip/hip_bf16.h>
#include <stdint.h>

// ---------------------------------------------------------------------------
// MI455X (gfx1250) implementation.
//   K0: W[256x256] f32 -> Wt[k][n] f16 (transposed, GEMM-B friendly)
//   K1: per-token quantum attention (wave32: lane == head), f16 output -> ws
//   K2: [65536x256] x [256x256] GEMM via v_wmma_f32_16x16x32_f16,
//       B-operand slice staged in LDS by the Tensor Data Mover (TDM),
//       f32 accumulate, +bias epilogue. B fragments preloaded per K-step so
//       the 4 WMMAs issue back-to-back behind a single s_wait_dscnt.
// ---------------------------------------------------------------------------

typedef __attribute__((ext_vector_type(16))) _Float16 v16h;
typedef __attribute__((ext_vector_type(8)))  _Float16 v8h;
typedef __attribute__((ext_vector_type(8)))  float    v8f;
typedef __attribute__((ext_vector_type(4)))  float    v4f;
typedef __attribute__((ext_vector_type(4)))  uint32_t v4u;
typedef __attribute__((ext_vector_type(8)))  int      v8i;
typedef __attribute__((ext_vector_type(4)))  int      v4i;

#define NHEADS 32
#define DK     8
#define EDIM   256
#define NTOK   (16 * 4096)

// --------------------------- K0: W -> Wt (f16, transposed) -----------------
__global__ __launch_bounds__(256) void wprep_kernel(const float* __restrict__ W,
                                                    _Float16* __restrict__ Wt) {
    // block = output row k of Wt (coalesced writes), thread = n.
    int k = blockIdx.x;
    int n = threadIdx.x;
    Wt[(size_t)k * EDIM + n] = (_Float16)W[(size_t)n * EDIM + k];
}

// --------------------------- K1: quantum attention -------------------------
// One wave32 per token; lane == head. Q (32x8) staged in LDS, online softmax.
__global__ __launch_bounds__(256) void qattn_kernel(const float* __restrict__ x,
                                                    const float* __restrict__ theta,
                                                    _Float16* __restrict__ Aout) {
    __shared__ __align__(16) float qs[8][NHEADS][DK];
    int wave = threadIdx.x >> 5;
    int lane = threadIdx.x & 31;
    size_t token = (size_t)blockIdx.x * 8 + wave;

    const float* xp = x + token * EDIM + (size_t)lane * DK;
    float q[DK];
#pragma unroll
    for (int d = 0; d < DK; ++d) q[d] = __cosf(xp[d] + theta[d]);
#pragma unroll
    for (int d = 0; d < DK; ++d) qs[wave][lane][d] = q[d];
    __syncthreads();

    const float inv_sqrt_dk = 0.35355339059327373f; // 1/sqrt(8)
    float m = -3.0e38f, l = 0.0f;
    float acc[DK] = {0.f, 0.f, 0.f, 0.f, 0.f, 0.f, 0.f, 0.f};

    for (int g = 0; g < NHEADS; ++g) {
        // broadcast read of head g's 8 values (uniform LDS address)
        v4f qg0 = *(const v4f*)&qs[wave][g][0];
        v4f qg1 = *(const v4f*)&qs[wave][g][4];
        float qg[DK] = {qg0[0], qg0[1], qg0[2], qg0[3],
                        qg1[0], qg1[1], qg1[2], qg1[3]};
        float s = 0.f;
#pragma unroll
        for (int d = 0; d < DK; ++d) s += q[d] * qg[d];
        s *= inv_sqrt_dk;
        float mn   = fmaxf(m, s);
        float corr = __expf(m - mn);
        float p    = __expf(s - mn);
        l = l * corr + p;
#pragma unroll
        for (int d = 0; d < DK; ++d) acc[d] = acc[d] * corr + p * qg[d];
        m = mn;
    }
    float invl = 1.0f / l;
    v8h o;
#pragma unroll
    for (int d = 0; d < DK; ++d) o[d] = (_Float16)(acc[d] * invl);
    *(v8h*)(Aout + token * EDIM + (size_t)lane * DK) = o;
}

// --------------------------- K2: WMMA GEMM ---------------------------------
// out[M=65536, N=256] = A[M,256] * Wt (+bias), Wt stored as [k][n].
// Workgroup tile: 128(M) x 64(N); 8 waves, each 16(M) x 64(N) (4 wmma N-tiles).
// Wt slice (K=256 x N=64, f16) loaded into LDS by TDM with row padding:
// LDS row stride = 64 + 8 = 72 f16 (144 B) to spread LDS banks.
#define LDS_ROW 72

__global__ __launch_bounds__(256) void qgemm_kernel(const _Float16* __restrict__ A,
                                                    const _Float16* __restrict__ Wt,
                                                    const float* __restrict__ bias,
                                                    float* __restrict__ out) {
    __shared__ __align__(16) _Float16 ldsW[256 * LDS_ROW]; // 36 KB

    int wave  = threadIdx.x >> 5;
    int lane  = threadIdx.x & 31;
    int mBase = blockIdx.x * 128 + wave * 16;
    int nBase = blockIdx.y * 64;

    if (wave == 0) {
        // ---- Tensor DMA descriptor (D#): 2D tile of Wt into LDS ----
        uint64_t ga = (uint64_t)(const void*)(Wt + nBase); // tile start
        uint32_t la = (uint32_t)(size_t)(&ldsW[0]);        // LDS byte offset
        v4u g0;
        g0[0] = 1u;                                    // count=1, user mode
        g0[1] = la;                                    // lds_addr
        g0[2] = (uint32_t)ga;                          // global_addr[31:0]
        g0[3] = ((uint32_t)(ga >> 32) & 0x01FFFFFFu)   // global_addr[56:32]
              | (2u << 30);                            // type=2 ("image")
        v8i g1;
        g1[0] = (int)((1u << 16)      // data_size = 1 -> 2-byte elements
              |       (1u << 20)      // pad_enable (LDS row padding)
              |       (4u << 22)      // pad_interval: 2^(4+1)=32 dwords (=128B row)
              |       (3u << 25));    // pad_amount: 3+1 = 4 dwords (=8 f16)
        g1[1] = (int)(256u << 16);    // tensor_dim0 = 256 (f16 per Wt row)
        g1[2] = (int)(256u << 16);    // tensor_dim0 hi=0 | tensor_dim1 = 256 rows
        g1[3] = (int)(64u  << 16);    // tensor_dim1 hi=0 | tile_dim0 = 64 f16
        g1[4] = (int)256;             // tile_dim1 = 256 rows, tile_dim2 = 0
        g1[5] = (int)256;             // tensor_dim0_stride = 256 (f16 units)
        g1[6] = 0;                    // stride hi | tensor_dim1_stride lo
        g1[7] = 0;
        v4i gz4 = {0, 0, 0, 0};
        v8i gz8 = {0, 0, 0, 0, 0, 0, 0, 0};
        // 6-arg form (clang-23 / therock-10.0 headers): (g0, g1, g2, g3, g4, cpol)
        __builtin_amdgcn_tensor_load_to_lds(g0, g1, gz4, gz4, gz8, 0);
        __builtin_amdgcn_s_wait_tensorcnt(0);
    }
    __syncthreads();

    // A operand: 16-bit A layout -> lanes 0-15: rows M=0..15, K 0..7/16..23;
    // lanes 16-31: same rows, K 8..15/24..31.
    const _Float16* Ap = A + (size_t)(mBase + (lane & 15)) * EDIM + (lane >> 4) * 8;
    // B operand: lane&15 == N; elements over K (table order: K 0..7/16..23 for lanes 0-15, 8..15/24..31 for lanes 16-31).
    const _Float16* Bbase = ldsW + (size_t)((lane >> 4) * 8) * LDS_ROW + (lane & 15);

    v8f acc[4];
#pragma unroll
    for (int t = 0; t < 4; ++t)
#pragma unroll
        for (int r = 0; r < 8; ++r) acc[t][r] = 0.f;

#pragma unroll
    for (int kk = 0; kk < 256; kk += 32) {
        union { v16h v; v8h h[2]; } a;
        a.h[0] = *(const v8h*)(Ap + kk);        // global_load_b128
        a.h[1] = *(const v8h*)(Ap + kk + 16);   // global_load_b128

        // Preload all 4 B fragments (64 strided 16-bit LDS reads per lane, single wait),
        // then issue 4 WMMAs back-to-back.
        union { v16h v; v8h h[2]; } b[4];
#pragma unroll
        for (int t = 0; t < 4; ++t) {
            const _Float16* Bp = Bbase + (size_t)kk * LDS_ROW + t * 16;
#pragma unroll
            for (int s = 0; s < 16; ++s)
                b[t].v[s] = Bp[(size_t)((s >> 3) * 16 + (s & 7)) * LDS_ROW];
        }
#pragma unroll
        for (int t = 0; t < 4; ++t) {
            acc[t] = __builtin_amdgcn_wmma_f32_16x16x32_f16(
                false, a.v, false, b[t].v, (short)0, acc[t], false, false);
        }
    }

    // Epilogue: C/D layout -> VGPR r: lanes 0-15 M=r, lanes 16-31 M=r+8; N=lane&15.
    int rowHalf = (lane >> 4) * 8;
    int col0    = lane & 15;
#pragma unroll
    for (int t = 0; t < 4; ++t) {
        int col  = nBase + t * 16 + col0;
        float bc = bias[col];
#pragma unroll
        for (int r = 0; r < 8; ++r) {
            int row = mBase + rowHalf + r;
            out[(size_t)row * EDIM + col] = acc[t][r] + bc;
        }
    }
}

// --------------------------- host launcher ---------------------------------
extern "C" void kernel_launch(void* const* d_in, const int* in_sizes, int n_in,
                              void* d_out, int out_size, void* d_ws, size_t ws_size,
                              hipStream_t stream) {
    const float* x     = (const float*)d_in[0];   // [16,4096,256]
    const float* theta = (const float*)d_in[1];   // [8]
    const float* W     = (const float*)d_in[2];   // [256,256]
    const float* bias  = (const float*)d_in[3];   // [256]
    float* out = (float*)d_out;                   // [16,4096,256]

    _Float16* Wt = (_Float16*)d_ws;                               // 128 KB
    _Float16* Aq = (_Float16*)((char*)d_ws + EDIM * EDIM * 2);    // 32 MB

    wprep_kernel<<<EDIM, EDIM, 0, stream>>>(W, Wt);
    qattn_kernel<<<NTOK / 8, 256, 0, stream>>>(x, theta, Aq);
    qgemm_kernel<<<dim3(NTOK / 128, EDIM / 64), 256, 0, stream>>>(Aq, Wt, bias, out);
}
